// Branch_4466765988603
// MI455X (gfx1250) — matrix operand order repaired and checked by reading, not compiled, not run on hardware
//
#include <hip/hip_runtime.h>

// Problem constants (match reference)
#define B_TOT 16384
#define DIN   512
#define HDIM  1024
#define DOUT  512
#define NEXP  4

typedef __bf16 v16bf __attribute__((ext_vector_type(16)));
typedef float  v8f   __attribute__((ext_vector_type(8)));

union BFrag { v16bf v; uint4 u[2]; };

// round-to-nearest-even float -> bf16 (as raw ushort)
static __device__ inline unsigned short f2bf(float f) {
    unsigned u = __float_as_uint(f);
    unsigned r = (u + 0x7FFFu + ((u >> 16) & 1u)) >> 16;
    return (unsigned short)r;
}
static __device__ inline float bf2f(unsigned short h) {
    return __uint_as_float(((unsigned)h) << 16);
}

__global__ void zero_counts(int* counts) {
    if (threadIdx.x < NEXP) counts[threadIdx.x] = 0;
}

// Bucket rows by expert. Order within a bucket is nondeterministic but every
// row's output value is order-independent (same expert, same math).
__global__ void scatter_kernel(const int* __restrict__ cmd, int* counts,
                               int* __restrict__ idxList) {
    int b = blockIdx.x * 256 + threadIdx.x;
    if (b < B_TOT) {
        int e = cmd[b] & (NEXP - 1);
        int p = atomicAdd(&counts[e], 1);
        idxList[e * B_TOT + p] = b;
    }
}

// W[e][K][N] fp32  ->  Th/Tl[e][N][K] bf16 hi / lo (N-major so B-fragments are
// contiguous K-runs per output column, matching the WMMA B VGPR layout).
__global__ __launch_bounds__(256)
void split_transpose(const float* __restrict__ W,
                     unsigned short* __restrict__ Th,
                     unsigned short* __restrict__ Tl, int K, int N) {
    __shared__ float tile[32][33];
    int e = blockIdx.z;
    int n0 = blockIdx.x * 32, k0 = blockIdx.y * 32;
    int tx = threadIdx.x, ty = threadIdx.y;   // 32 x 8
    const float* src = W + ((size_t)e * K + k0) * N + n0;
#pragma unroll
    for (int i = 0; i < 32; i += 8)
        tile[ty + i][tx] = src[(size_t)(ty + i) * N + tx];
    __syncthreads();
    size_t obase = ((size_t)e * N + n0) * K + k0;
#pragma unroll
    for (int i = 0; i < 32; i += 8) {
        float f = tile[tx][ty + i];
        unsigned short hi = f2bf(f);
        unsigned short lo = f2bf(f - bf2f(hi));
        Th[obase + (size_t)(ty + i) * K + tx] = hi;
        Tl[obase + (size_t)(ty + i) * K + tx] = lo;
    }
}

// Grouped GEMM: C[rows(e), NT] = act(A @ W[e]^T-major + bias[e]).
// Block = 256 thr = 8 waves, tile 64(M) x 128(N); wave = 32x32 (2x2 of 16x16).
// bf16x3 split accumulation in f32 via v_wmma_f32_16x16x32_bf16.
template <bool A_FP32, bool FINAL, int K, int NT>
__global__ __launch_bounds__(256)
void gemm_kernel(const float* __restrict__ xA,
                 const unsigned short* __restrict__ Ahg,
                 const unsigned short* __restrict__ Alg,
                 const unsigned short* __restrict__ Wh,
                 const unsigned short* __restrict__ Wl,
                 const float* __restrict__ bias,
                 const int* __restrict__ counts,
                 const int* __restrict__ idxList,
                 unsigned short* __restrict__ Oh,
                 unsigned short* __restrict__ Ol,
                 float* __restrict__ Of) {
    // map blockIdx.x -> (expert e, m-tile mt); uniform across the block
    int e = -1, mt = 0, acc = 0;
#pragma unroll
    for (int i = 0; i < NEXP; ++i) {
        int c = counts[i];
        int t = (c + 63) >> 6;
        if (e < 0 && (int)blockIdx.x < acc + t) { e = i; mt = (int)blockIdx.x - acc; }
        acc += t;
    }
    if (e < 0) return;
    int cnt = counts[e];

    __shared__ __align__(16) unsigned short AhS[64][72];  // pad vs bank conflicts
    __shared__ __align__(16) unsigned short AlS[64][72];
    __shared__ int rowS[64];

    int tid = threadIdx.x;
    if (tid < 64) {
        int s = mt * 64 + tid;
        int sc = s < cnt ? s : cnt - 1;   // clamp: pad slots load a valid row
        rowS[tid] = idxList[e * B_TOT + sc];
    }
    __syncthreads();

    int lane = tid & 31, w = tid >> 5;
    int wm = w >> 2, wn = w & 3;          // 2 x 4 wave grid
    int nb = (int)blockIdx.y * 128;

    int lr = tid >> 2;                    // staging: row 0..63
    int lk = (tid & 3) * 16;              // staging: 16-elem K run
    int rowL = rowS[lr];

    v8f accf[2][2] = {};

    for (int kb = 0; kb < K; kb += 64) {
        // ---- stage A block 64 x 64 into LDS as bf16 hi/lo ----
        if (A_FP32) {
            const float* src = xA + (size_t)rowL * K + kb + lk;
#pragma unroll
            for (int i = 0; i < 16; ++i) {
                float f = src[i];
                unsigned short hi = f2bf(f);
                AhS[lr][lk + i] = hi;
                AlS[lr][lk + i] = f2bf(f - bf2f(hi));
            }
        } else {
            const uint4* sh = (const uint4*)(Ahg + (size_t)rowL * K + kb + lk);
            const uint4* sl = (const uint4*)(Alg + (size_t)rowL * K + kb + lk);
            *(uint4*)&AhS[lr][lk]     = sh[0];
            *(uint4*)&AhS[lr][lk + 8] = sh[1];
            *(uint4*)&AlS[lr][lk]     = sl[0];
            *(uint4*)&AlS[lr][lk + 8] = sl[1];
        }
        __syncthreads();

#pragma unroll
        for (int ks = 0; ks < 64; ks += 32) {
            BFrag ah[2], al[2], bh[2], bl[2];
            // A fragment, contiguous K order (k = 16h + s): lanes 0-15 = K lo-halves,
            // lanes 16-31 = K hi-halves of the same rows.
            int arow0 = wm * 32 + (lane & 15);
            int o1 = ks + ((lane >> 4) << 4);
#pragma unroll
            for (int mi = 0; mi < 2; ++mi) {
                int arow = arow0 + mi * 16;
                ah[mi].u[0] = *(const uint4*)&AhS[arow][o1];
                ah[mi].u[1] = *(const uint4*)&AhS[arow][o1 + 8];
                al[mi].u[0] = *(const uint4*)&AlS[arow][o1];
                al[mi].u[1] = *(const uint4*)&AlS[arow][o1 + 8];
            }
            // B fragment: column (lane&15), contiguous K run (N-major weights);
            // lanes 16-31 take K+16..31.  Streams from L2 (32MB weights fit).
            int kg = kb + ks + ((lane >> 4) << 4);
#pragma unroll
            for (int ni = 0; ni < 2; ++ni) {
                int col = nb + wn * 32 + ni * 16 + (lane & 15);
                const uint4* ph = (const uint4*)(Wh + ((size_t)e * NT + col) * K + kg);
                const uint4* pl = (const uint4*)(Wl + ((size_t)e * NT + col) * K + kg);
                bh[ni].u[0] = ph[0]; bh[ni].u[1] = ph[1];
                bl[ni].u[0] = pl[0]; bl[ni].u[1] = pl[1];
            }
#pragma unroll
            for (int mi = 0; mi < 2; ++mi)
#pragma unroll
                for (int ni = 0; ni < 2; ++ni) {
                    accf[mi][ni] = __builtin_amdgcn_wmma_f32_16x16x32_bf16(
                        false, ah[mi].v, false, bh[ni].v, (short)0, accf[mi][ni], false, false);
                    accf[mi][ni] = __builtin_amdgcn_wmma_f32_16x16x32_bf16(
                        false, ah[mi].v, false, bl[ni].v, (short)0, accf[mi][ni], false, false);
                    accf[mi][ni] = __builtin_amdgcn_wmma_f32_16x16x32_bf16(
                        false, al[mi].v, false, bh[ni].v, (short)0, accf[mi][ni], false, false);
                }
        }
        __syncthreads();
    }

    // ---- epilogue: bias + activation; C layout lane(&15)=N, vgpr r=M (+8 hi) ----
#pragma unroll
    for (int ni = 0; ni < 2; ++ni) {
        int ncol = nb + wn * 32 + ni * 16 + (lane & 15);
        float bv = bias[(size_t)e * NT + ncol];
#pragma unroll
        for (int mi = 0; mi < 2; ++mi) {
#pragma unroll
            for (int r = 0; r < 8; ++r) {
                int ls = wm * 32 + mi * 16 + r + ((lane >> 4) << 3);
                int s = mt * 64 + ls;
                if (s < cnt) {
                    int row = rowS[ls];
                    float f = accf[mi][ni][r] + bv;
                    if (FINAL) {
                        Of[(size_t)row * NT + ncol] = tanhf(f);
                    } else {
                        f = fmaxf(f, 0.0f);
                        unsigned short hi = f2bf(f);
                        unsigned short lo = f2bf(f - bf2f(hi));
                        Oh[(size_t)row * NT + ncol] = hi;
                        Ol[(size_t)row * NT + ncol] = lo;
                    }
                }
            }
        }
    }
}

extern "C" void kernel_launch(void* const* d_in, const int* in_sizes, int n_in,
                              void* d_out, int out_size, void* d_ws, size_t ws_size,
                              hipStream_t stream) {
    const float* x  = (const float*)d_in[0];
    const int*   cmd = (const int*)d_in[1];   // command (int32 per harness convention)
    const float* W1 = (const float*)d_in[2];
    const float* b1 = (const float*)d_in[3];
    const float* W2 = (const float*)d_in[4];
    const float* b2 = (const float*)d_in[5];
    const float* W3 = (const float*)d_in[6];
    const float* b3 = (const float*)d_in[7];
    float* out = (float*)d_out;

    char* ws = (char*)d_ws;
    size_t off = 0;
    int* counts  = (int*)(ws + off); off += 256;
    int* idxList = (int*)(ws + off); off += (size_t)NEXP * B_TOT * sizeof(int);
    const size_t w1sz = (size_t)NEXP * HDIM * DIN  * 2;  // bytes (2 per bf16)
    const size_t w2sz = (size_t)NEXP * HDIM * HDIM * 2;
    const size_t w3sz = (size_t)NEXP * DOUT * HDIM * 2;
    unsigned short* W1h = (unsigned short*)(ws + off); off += w1sz;
    unsigned short* W1l = (unsigned short*)(ws + off); off += w1sz;
    unsigned short* W2h = (unsigned short*)(ws + off); off += w2sz;
    unsigned short* W2l = (unsigned short*)(ws + off); off += w2sz;
    unsigned short* W3h = (unsigned short*)(ws + off); off += w3sz;
    unsigned short* W3l = (unsigned short*)(ws + off); off += w3sz;
    const size_t hsz = (size_t)B_TOT * HDIM * 2;
    unsigned short* h1h = (unsigned short*)(ws + off); off += hsz;
    unsigned short* h1l = (unsigned short*)(ws + off); off += hsz;
    unsigned short* h2h = (unsigned short*)(ws + off); off += hsz;
    unsigned short* h2l = (unsigned short*)(ws + off); off += hsz;
    (void)in_sizes; (void)n_in; (void)out_size; (void)ws_size;

    zero_counts<<<1, 32, 0, stream>>>(counts);
    scatter_kernel<<<B_TOT / 256, 256, 0, stream>>>(cmd, counts, idxList);

    split_transpose<<<dim3(HDIM / 32, DIN / 32,  NEXP), dim3(32, 8), 0, stream>>>(W1, W1h, W1l, DIN,  HDIM);
    split_transpose<<<dim3(HDIM / 32, HDIM / 32, NEXP), dim3(32, 8), 0, stream>>>(W2, W2h, W2l, HDIM, HDIM);
    split_transpose<<<dim3(DOUT / 32, HDIM / 32, NEXP), dim3(32, 8), 0, stream>>>(W3, W3h, W3l, HDIM, DOUT);

    dim3 g1(B_TOT / 64 + NEXP, HDIM / 128);
    gemm_kernel<true,  false, DIN,  HDIM><<<g1, 256, 0, stream>>>(
        x, nullptr, nullptr, W1h, W1l, b1, counts, idxList, h1h, h1l, nullptr);
    dim3 g2(B_TOT / 64 + NEXP, HDIM / 128);
    gemm_kernel<false, false, HDIM, HDIM><<<g2, 256, 0, stream>>>(
        nullptr, h1h, h1l, W2h, W2l, b2, counts, idxList, h2h, h2l, nullptr);
    dim3 g3(B_TOT / 64 + NEXP, DOUT / 128);
    gemm_kernel<false, true,  HDIM, DOUT><<<g3, 256, 0, stream>>>(
        nullptr, h2h, h2l, W3h, W3l, b3, counts, idxList, nullptr, nullptr, out);
}
